// LSTMTagger_46669114638388
// MI455X (gfx1250) — matrix operand order repaired and checked by reading, not compiled, not run on hardware
//
#include <hip/hip_runtime.h>
#include <hip/hip_bf16.h>
#include <math.h>

// ---------------- problem constants (from reference) ----------------
#define SEQn      8192
#define MAXW      16
#define CHDIM     32      // CHAR_DIM
#define NDIM      256     // N_DIM
#define CHID      64      // CHAR_HIDDEN
#define NHID      256     // N_HIDDEN
#define NTAG      50
#define NTAGP     64      // padded tag count
#define FEATD     320     // NDIM + CHID
#define CGATES    256     // 4*CHID
#define WGATES    1024    // 4*NHID
#define NSEQWG    16      // workgroups in persistent sequential kernel
#define HPAD      72      // padded LDS row stride for char h (bank-conflict-free)

typedef __attribute__((ext_vector_type(16))) __bf16       v16bf;
typedef __attribute__((ext_vector_type(8)))  float        v8f;
typedef __attribute__((ext_vector_type(8)))  unsigned int v8u;

// ---------------- scalar helpers ----------------
__device__ __forceinline__ __bf16 f2bf(float f) {
  unsigned u = __builtin_bit_cast(unsigned, f);
  unsigned r = u + 0x7FFFu + ((u >> 16) & 1u);       // round-to-nearest-even
  return __builtin_bit_cast(__bf16, (unsigned short)(r >> 16));
}
__device__ __forceinline__ float bfu2f(unsigned short u) {
  return __builtin_bit_cast(float, ((unsigned)u) << 16);
}
__device__ __forceinline__ float sigm(float x) { return 1.0f / (1.0f + expf(-x)); }

// ---------------- WMMA fragment builders (contiguous-K order, same for A and B) ----
// All sources are pre-converted bf16, so these are pure packed loads.
// A (16x32 bf16, MxK): lane L holds row M=L&15 (rowp = that row's base).
//   lanes 0-15 : K=0..15 ; lanes 16-31: K=16..31
__device__ __forceinline__ v16bf afrag_bf16(const __bf16* rowp, int kbase) {
  const int lane = threadIdx.x & 31;
  const int hi8  = (lane >> 4) * 16;
  const unsigned int* p = (const unsigned int*)(rowp + kbase + hi8);
  v8u u;
#pragma unroll
  for (int v = 0; v < 4; ++v) u[v] = p[v];         // K = hi8 + 0..7
#pragma unroll
  for (int v = 0; v < 4; ++v) u[4 + v] = p[4 + v]; // K = hi8 + 8..15
  return __builtin_bit_cast(v16bf, u);
}
// B (32x16 bf16, KxN) from row-major bf16 weight Wb[nrows][ldk]
// (B[k][n] = Wb[n0+n][kbase+k]):
//   lanes 0-15 : col N=lane, K=0..15 ; lanes 16-31: col N=lane-16, K=16..31
__device__ __forceinline__ v16bf bfrag_bf16(const __bf16* Wb, int ldk, int n0,
                                            int kbase) {
  const int lane = threadIdx.x & 31;
  const int col  = lane & 15;
  const int khi  = (lane >> 4) * 16;
  const unsigned int* p =
      (const unsigned int*)(Wb + (size_t)(n0 + col) * ldk + kbase + khi);
  v8u u;
#pragma unroll
  for (int v = 0; v < 8; ++v) u[v] = p[v];         // 8 consecutive dwords
  return __builtin_bit_cast(v16bf, u);
}
#define WMMA_BF16(A, B, C) \
  __builtin_amdgcn_wmma_f32_16x16x32_bf16(false, (A), false, (B), (short)0, (C), false, false)

// ---------------- prep kernels (run once per launch) ----------------
__global__ void cvt_kernel(const float* __restrict__ s, __bf16* d, int n) {
  int i = blockIdx.x * blockDim.x + threadIdx.x;
  if (i < n) d[i] = f2bf(s[i]);
}
__global__ void cvt_linW_kernel(const float* __restrict__ w, __bf16* d) {
  int i = blockIdx.x * blockDim.x + threadIdx.x;  // over NTAGP*NHID
  if (i < NTAGP * NHID) {
    int n = i >> 8, k = i & 255;
    d[i] = (n < NTAG) ? f2bf(w[n * NHID + k])
                      : __builtin_bit_cast(__bf16, (unsigned short)0);
  }
}
__global__ void bias_kernel(const float* __restrict__ a, const float* __restrict__ b,
                            float* d, int n) {
  int i = blockIdx.x * blockDim.x + threadIdx.x;
  if (i < n) d[i] = a[i] + b[i];
}
__global__ void init_kernel(float* hbuf, unsigned* sync) {
  int i = blockIdx.x * blockDim.x + threadIdx.x;
  if (i < 2 * NHID) hbuf[i] = 0.0f;
  if (i == 0) *sync = 0u;
}

// ---------------- char LSTM: all 16 timesteps in one kernel --------------
// block = 16 words, 256 threads (8 waves); wave w owns gate cols [32w,32w+32).
// Per-word recurrence is block-local -> only __syncthreads between steps.
// h (bf16, padded rows) and c (f32) live in LDS; weight B-fragments are
// hoisted into registers across the whole time loop.
__global__ __launch_bounds__(256) void char_lstm_kernel(
    const int* __restrict__ char_idx, const int* __restrict__ char_lens,
    const __bf16* __restrict__ char_emb_b, const __bf16* __restrict__ c_Wih_b,
    const __bf16* __restrict__ c_Whh_b, const float* __restrict__ b_c,
    __bf16* h_bf_out) {
  __shared__ float  gsm[16][CGATES];          // gates, 16KB
  __shared__ __bf16 hsm[16][HPAD];            // h state (padded), 2.25KB
  __shared__ float  csm[16][CHID];            // c state, 4KB
  const int s0   = blockIdx.x * 16;
  const int wave = threadIdx.x >> 5;
  const int lane = threadIdx.x & 31;
  const int hi   = lane >> 4;
  const int row  = lane & 15;
  const int n0   = wave * 32;

  // zero-init block-local state
  for (int it = threadIdx.x; it < 16 * CHID; it += 256) {
    int mm = it >> 6, j = it & 63;
    hsm[mm][j] = __builtin_bit_cast(__bf16, (unsigned short)0);
    csm[mm][j] = 0.0f;
  }

  // time-invariant weight B-fragments (2 N-tiles x {Wih, Whh lo, Whh hi})
  v16bf bx0  = bfrag_bf16(c_Wih_b, CHDIM, n0, 0);
  v16bf bh00 = bfrag_bf16(c_Whh_b, CHID, n0, 0);
  v16bf bh01 = bfrag_bf16(c_Whh_b, CHID, n0, 32);
  v16bf bx1  = bfrag_bf16(c_Wih_b, CHDIM, n0 + 16, 0);
  v16bf bh10 = bfrag_bf16(c_Whh_b, CHID, n0 + 16, 0);
  v16bf bh11 = bfrag_bf16(c_Whh_b, CHID, n0 + 16, 32);
  __syncthreads();

  for (int t = 0; t < MAXW; ++t) {
    // A fragments: gathered char embedding (K=32) + LDS-resident h (K=64)
    const int ci       = char_idx[(s0 + row) * MAXW + t];
    const __bf16* xrow = char_emb_b + (size_t)ci * CHDIM;
    v16bf ax  = afrag_bf16(xrow, 0);
    v16bf ah0 = afrag_bf16(&hsm[row][0], 0);
    v16bf ah1 = afrag_bf16(&hsm[row][0], 32);

    v8f acc0 = {}, acc1 = {};
    acc0 = WMMA_BF16(ax, bx0, acc0);
    acc0 = WMMA_BF16(ah0, bh00, acc0);
    acc0 = WMMA_BF16(ah1, bh01, acc0);
    acc1 = WMMA_BF16(ax, bx1, acc1);
    acc1 = WMMA_BF16(ah0, bh10, acc1);
    acc1 = WMMA_BF16(ah1, bh11, acc1);
#pragma unroll
    for (int r = 0; r < 8; ++r) {
      int mm = r + 8 * hi;
      int nn = lane & 15;
      gsm[mm][n0 + nn]      = acc0[r] + b_c[n0 + nn];
      gsm[mm][n0 + 16 + nn] = acc1[r] + b_c[n0 + 16 + nn];
    }
    __syncthreads();

    // gate nonlinearity + masked state update: 16 words x 64 hidden
    for (int it = threadIdx.x; it < 16 * CHID; it += 256) {
      int mm = it >> 6, j = it & 63;
      if (t < char_lens[s0 + mm]) {
        float iv = sigm(gsm[mm][j]);
        float fv = sigm(gsm[mm][CHID + j]);
        float gv = tanhf(gsm[mm][2 * CHID + j]);
        float ov = sigm(gsm[mm][3 * CHID + j]);
        float cn = fv * csm[mm][j] + iv * gv;
        float hn = ov * tanhf(cn);
        csm[mm][j] = cn;
        hsm[mm][j] = f2bf(hn);
      }
    }
    __syncthreads();
  }

  // final char h -> global (bf16 is all downstream consumers need)
  for (int it = threadIdx.x; it < 16 * CHID; it += 256) {
    int mm = it >> 6, j = it & 63;
    h_bf_out[(size_t)(s0 + mm) * CHID + j] = hsm[mm][j];
  }
}

// ---------------- build feats = [word_emb[x[s]], char_h[s]] as bf16 -------
__global__ __launch_bounds__(256) void feats_kernel(const int* __restrict__ x,
                                                    const float* __restrict__ word_emb,
                                                    const __bf16* __restrict__ h_bf,
                                                    __bf16* feats) {
  int s = blockIdx.x;
  const float* we = word_emb + (size_t)x[s] * NDIM;
  for (int k = threadIdx.x; k < FEATD; k += 256) {
    __bf16 v = (k < NDIM) ? f2bf(we[k]) : h_bf[(size_t)s * CHID + (k - NDIM)];
    feats[(size_t)s * FEATD + k] = v;
  }
}

// ---------------- word LSTM input projection: feats @ Wih.T + b (WMMA) ----
// grid (512, 16), 128 threads (4 waves); wave owns one 16-wide N tile
__global__ __launch_bounds__(128) void wordx_gemm_kernel(
    const __bf16* __restrict__ feats, const __bf16* __restrict__ w_Wih_b,
    const float* __restrict__ b_w, float* gates_x) {
  const int s0   = blockIdx.x * 16;
  const int wave = threadIdx.x >> 5;
  const int lane = threadIdx.x & 31;
  const int hi   = lane >> 4;
  const int n0   = blockIdx.y * 64 + wave * 16;
  const __bf16* arow = feats + (size_t)(s0 + (lane & 15)) * FEATD;
  v8f acc = {};
#pragma unroll
  for (int kc = 0; kc < FEATD; kc += 32) {
    v16bf a = afrag_bf16(arow, kc);
    v16bf b = bfrag_bf16(w_Wih_b, FEATD, n0, kc);
    acc = WMMA_BF16(a, b, acc);
  }
#pragma unroll
  for (int r = 0; r < 8; ++r) {
    int mm = r + 8 * hi;
    int ng = n0 + (lane & 15);
    gates_x[(size_t)(s0 + mm) * WGATES + ng] = acc[r] + b_w[ng];
  }
}

// ---------------- persistent sequential word LSTM -------------------------
// 16 WGs x 64 threads. WG wg owns hidden slice [16wg,16wg+16); thread tid:
// gate g=tid>>4, local hidden jj=tid&15, Whh row = 256g + 16wg + jj.
// Whh slice lives in LDS (bf16, transposed, 32KB). Global spin barrier/step.
__global__ __launch_bounds__(64) void word_seq_kernel(
    const float* __restrict__ gates_x, const __bf16* __restrict__ w_Whh_b,
    float* hbuf /*2 x 256, double buffered*/, __bf16* hs_bf, unsigned* sync) {
  __shared__ unsigned short wsm[NHID][64];  // [k][row_local] bf16 bits, 32KB
  __shared__ float hsm[NHID];
  __shared__ float gsm[4][16];
  const int tid = threadIdx.x;
  const int wg  = blockIdx.x;
  const int g   = tid >> 4;
  const int jj  = tid & 15;
  const int row = g * NHID + wg * 16 + jj;

  // stage Whh row into LDS transposed (conflict-free on read)
  const unsigned short* wrow =
      (const unsigned short*)w_Whh_b + (size_t)row * NHID;
  for (int k = 0; k < NHID; ++k) wsm[k][tid] = wrow[k];
  float creg = 0.0f;  // persistent cell state (meaningful for tid < 16)
  __syncthreads();

  for (int step = 0; step < SEQn; ++step) {
    // hide next step's gates_x fetch behind this step's dot product
    if (step + 1 < SEQn)
      __builtin_prefetch(&gates_x[(size_t)(step + 1) * WGATES + row], 0, 1);

    const float* hin = hbuf + (size_t)(step & 1) * NHID;
    for (int k = tid; k < NHID; k += 64) hsm[k] = hin[k];
    __syncthreads();

    float a0 = 0.f, a1 = 0.f, a2 = 0.f, a3 = 0.f;
#pragma unroll
    for (int k = 0; k < NHID; k += 4) {
      a0 += bfu2f(wsm[k + 0][tid]) * hsm[k + 0];
      a1 += bfu2f(wsm[k + 1][tid]) * hsm[k + 1];
      a2 += bfu2f(wsm[k + 2][tid]) * hsm[k + 2];
      a3 += bfu2f(wsm[k + 3][tid]) * hsm[k + 3];
    }
    gsm[g][jj] = gates_x[(size_t)step * WGATES + row] + (a0 + a1) + (a2 + a3);
    __syncthreads();

    if (tid < 16) {
      float iv = sigm(gsm[0][tid]);
      float fv = sigm(gsm[1][tid]);
      float gv = tanhf(gsm[2][tid]);
      float ov = sigm(gsm[3][tid]);
      creg = fv * creg + iv * gv;
      float hv = ov * tanhf(creg);
      int j = wg * 16 + tid;
      hs_bf[(size_t)step * NHID + j]            = f2bf(hv);
      hbuf[(size_t)((step + 1) & 1) * NHID + j] = hv;  // write opposite buffer
    }
    __syncthreads();
    __threadfence();
    if (tid == 0) {
      __hip_atomic_fetch_add(sync, 1u, __ATOMIC_ACQ_REL, __HIP_MEMORY_SCOPE_AGENT);
      unsigned target = (unsigned)NSEQWG * (unsigned)(step + 1);
      while (__hip_atomic_load(sync, __ATOMIC_ACQUIRE, __HIP_MEMORY_SCOPE_AGENT) <
             target) {
        __builtin_amdgcn_s_sleep(1);
      }
    }
    __syncthreads();
  }
}

// ---------------- logits (WMMA, N padded 50->64) + softmax ---------------
__global__ __launch_bounds__(32) void logits_softmax_kernel(
    const __bf16* __restrict__ hs_bf, const __bf16* __restrict__ lin_Wb,
    const float* __restrict__ lin_b, float* out) {
  __shared__ float lsm[16][NTAGP];
  const int s0   = blockIdx.x * 16;
  const int lane = threadIdx.x;
  const int hi   = lane >> 4;
  const __bf16* arow = hs_bf + (size_t)(s0 + (lane & 15)) * NHID;
  v8f acc0 = {}, acc1 = {}, acc2 = {}, acc3 = {};
#pragma unroll
  for (int kc = 0; kc < NHID; kc += 32) {
    v16bf a  = afrag_bf16(arow, kc);
    v16bf b0 = bfrag_bf16(lin_Wb, NHID, 0, kc);
    acc0 = WMMA_BF16(a, b0, acc0);
    v16bf b1 = bfrag_bf16(lin_Wb, NHID, 16, kc);
    acc1 = WMMA_BF16(a, b1, acc1);
    v16bf b2 = bfrag_bf16(lin_Wb, NHID, 32, kc);
    acc2 = WMMA_BF16(a, b2, acc2);
    v16bf b3 = bfrag_bf16(lin_Wb, NHID, 48, kc);
    acc3 = WMMA_BF16(a, b3, acc3);
  }
#pragma unroll
  for (int r = 0; r < 8; ++r) {
    int mm = r + 8 * hi, nn = lane & 15;
    lsm[mm][nn]      = acc0[r] + lin_b[nn];
    lsm[mm][16 + nn] = acc1[r] + lin_b[16 + nn];
    lsm[mm][32 + nn] = acc2[r] + lin_b[32 + nn];
    lsm[mm][48 + nn] = (48 + nn < NTAG) ? (acc3[r] + lin_b[48 + nn]) : -1e30f;
  }
  __syncthreads();
  if (lane < 16) {
    float mx = -1e30f;
    for (int n = 0; n < NTAG; ++n) mx = fmaxf(mx, lsm[lane][n]);
    float sum = 0.f;
    for (int n = 0; n < NTAG; ++n) {
      float e = expf(lsm[lane][n] - mx);
      lsm[lane][n] = e;
      sum += e;
    }
    float inv = 1.0f / sum;
    for (int n = 0; n < NTAG; ++n)
      out[(size_t)(s0 + lane) * NTAG + n] = lsm[lane][n] * inv;
  }
}

// ---------------- host-side orchestration ----------------
extern "C" void kernel_launch(void* const* d_in, const int* in_sizes, int n_in,
                              void* d_out, int out_size, void* d_ws, size_t ws_size,
                              hipStream_t stream) {
  const int*   x         = (const int*)d_in[0];
  const int*   char_idx  = (const int*)d_in[1];
  const int*   char_lens = (const int*)d_in[2];
  const float* word_emb  = (const float*)d_in[3];
  const float* char_emb  = (const float*)d_in[4];
  const float* c_Wih     = (const float*)d_in[5];
  const float* c_Whh     = (const float*)d_in[6];
  const float* c_bih     = (const float*)d_in[7];
  const float* c_bhh     = (const float*)d_in[8];
  const float* w_Wih     = (const float*)d_in[9];
  const float* w_Whh     = (const float*)d_in[10];
  const float* w_bih     = (const float*)d_in[11];
  const float* w_bhh     = (const float*)d_in[12];
  const float* lin_W     = (const float*)d_in[13];
  const float* lin_b     = (const float*)d_in[14];
  float*       out       = (float*)d_out;

  char* p = (char*)d_ws;
  auto alloc = [&](size_t bytes) -> void* {
    void* r = (void*)p;
    p += (bytes + 255) & ~(size_t)255;
    return r;
  };
  // state / activations
  __bf16*   h_bf    = (__bf16*)alloc((size_t)SEQn * CHID * 2);
  __bf16*   feats   = (__bf16*)alloc((size_t)SEQn * FEATD * 2);
  float*    gates_x = (float*)alloc((size_t)SEQn * WGATES * 4);
  float*    hbuf    = (float*)alloc(2 * NHID * 4);
  __bf16*   hs_bf   = (__bf16*)alloc((size_t)SEQn * NHID * 2);
  unsigned* sync    = (unsigned*)alloc(256);
  // pre-converted bf16 weights + combined biases
  __bf16* char_emb_b = (__bf16*)alloc((size_t)128 * CHDIM * 2);
  __bf16* c_Wih_b    = (__bf16*)alloc((size_t)CGATES * CHDIM * 2);
  __bf16* c_Whh_b    = (__bf16*)alloc((size_t)CGATES * CHID * 2);
  __bf16* w_Wih_b    = (__bf16*)alloc((size_t)WGATES * FEATD * 2);
  __bf16* w_Whh_b    = (__bf16*)alloc((size_t)WGATES * NHID * 2);
  __bf16* lin_Wb     = (__bf16*)alloc((size_t)NTAGP * NHID * 2);
  float*  b_c        = (float*)alloc(CGATES * 4);
  float*  b_w        = (float*)alloc(WGATES * 4);

  // ---- prep: zero sync state, convert weights once (graph-replay safe) ----
  init_kernel<<<2, 256, 0, stream>>>(hbuf, sync);
  cvt_kernel<<<(128 * CHDIM + 255) / 256, 256, 0, stream>>>(char_emb, char_emb_b, 128 * CHDIM);
  cvt_kernel<<<(CGATES * CHDIM + 255) / 256, 256, 0, stream>>>(c_Wih, c_Wih_b, CGATES * CHDIM);
  cvt_kernel<<<(CGATES * CHID + 255) / 256, 256, 0, stream>>>(c_Whh, c_Whh_b, CGATES * CHID);
  cvt_kernel<<<(WGATES * FEATD + 255) / 256, 256, 0, stream>>>(w_Wih, w_Wih_b, WGATES * FEATD);
  cvt_kernel<<<(WGATES * NHID + 255) / 256, 256, 0, stream>>>(w_Whh, w_Whh_b, WGATES * NHID);
  cvt_linW_kernel<<<(NTAGP * NHID + 255) / 256, 256, 0, stream>>>(lin_W, lin_Wb);
  bias_kernel<<<1, CGATES, 0, stream>>>(c_bih, c_bhh, b_c, CGATES);
  bias_kernel<<<4, 256, 0, stream>>>(w_bih, w_bhh, b_w, WGATES);

  // char LSTM: all 16 timesteps fused, state in LDS, weights in registers
  char_lstm_kernel<<<SEQn / 16, 256, 0, stream>>>(char_idx, char_lens, char_emb_b,
                                                  c_Wih_b, c_Whh_b, b_c, h_bf);
  // feats = [word_emb gather | char_h] in bf16
  feats_kernel<<<SEQn, 256, 0, stream>>>(x, word_emb, h_bf, feats);
  // time-independent input projection for word LSTM
  wordx_gemm_kernel<<<dim3(SEQn / 16, WGATES / 64), 128, 0, stream>>>(
      feats, w_Wih_b, b_w, gates_x);
  // sequential recurrence (persistent, LDS-resident Whh, global barrier/step)
  word_seq_kernel<<<NSEQWG, 64, 0, stream>>>(gates_x, w_Whh_b, hbuf, hs_bf, sync);
  // final linear (WMMA) + softmax
  logits_softmax_kernel<<<SEQn / 16, 32, 0, stream>>>(hs_bf, lin_Wb, lin_b, out);
}
